// MOE_12266426597652
// MI455X (gfx1250) — matrix operand order repaired and checked by reading, not compiled, not run on hardware
//
#include <hip/hip_runtime.h>
#include <hip/hip_bf16.h>
#include <cstdint>
#include <cstddef>

// Problem constants (match reference)
#define B_SZ   16384
#define S_SZ   512
#define H0G_SZ 512
#define H1G_SZ 256
#define H0E_SZ 1024
#define H1E_SZ 512
#define A_SZ   64
#define E_SZ   4

typedef __attribute__((ext_vector_type(16))) _Float16 v16h;
typedef __attribute__((ext_vector_type(8)))  _Float16 v8h;
typedef __attribute__((ext_vector_type(4)))  _Float16 v4h;
typedef __attribute__((ext_vector_type(8)))  float    v8f;
typedef __attribute__((ext_vector_type(4)))  float    v4f;

// ---------------------------------------------------------------------------
// WMMA helper: D = A(16x32 f16) * B(32x16 f16) + C(16x16 f32)
// ---------------------------------------------------------------------------
__device__ __forceinline__ v8f wmma_f16(v16h a, v16h b, v8f c) {
  return __builtin_amdgcn_wmma_f32_16x16x32_f16(
      false, a, false, b, (short)0, c, false, false);
}

// Fast tanh: 1 - 2/(exp2(2*log2e*x)+1).  ~5 VALU ops, exact at +-inf.
__device__ __forceinline__ float fast_tanh(float x) {
  float e = __builtin_amdgcn_exp2f(x * 2.8853900817779268f);
  return 1.0f - 2.0f * __builtin_amdgcn_rcpf(e + 1.0f);
}

// CDNA5 async global->LDS copy (ASYNCcnt-tracked), 16B per lane.
__device__ __forceinline__ void async_copy_b128(uint32_t lds_off, const void* gptr) {
  asm volatile("global_load_async_to_lds_b128 %0, %1, off"
               :: "v"(lds_off), "v"((uint64_t)(uintptr_t)gptr) : "memory");
}
__device__ __forceinline__ void wait_async0() {
  asm volatile("s_wait_asynccnt 0" ::: "memory");
}

// Fragment assembly from two 8-half loads (works for global and LDS pointers).
__device__ __forceinline__ v16h frag16_gap16(const _Float16* p) {   // A: halves at +0..7, +16..23
  v8h lo = *(const v8h*)p;
  v8h hi = *(const v8h*)(p + 16);
  return __builtin_shufflevector(lo, hi, 0,1,2,3,4,5,6,7,8,9,10,11,12,13,14,15);
}
__device__ __forceinline__ v16h frag16_cont(const _Float16* p) {    // B: 16 contiguous halves
  v8h lo = *(const v8h*)p;
  v8h hi = *(const v8h*)(p + 8);
  return __builtin_shufflevector(lo, hi, 0,1,2,3,4,5,6,7,8,9,10,11,12,13,14,15);
}

// Direct-from-global fragment loaders (used by the small combine kernel).
__device__ __forceinline__ v16h load_a_frag(const _Float16* __restrict__ X,
                                            int K, int row, int k0, int lane) {
  return frag16_cont(X + (size_t)row * K + k0 + ((lane >> 4) << 4));
}
__device__ __forceinline__ v16h load_b_frag(const _Float16* __restrict__ WT,
                                            int K, int col, int k0, int lane) {
  return frag16_cont(WT + (size_t)col * K + k0 + ((lane >> 4) << 4));
}

// ---------------------------------------------------------------------------
// Weight transpose+convert, tiled through LDS (K,N multiples of 32):
//   WT[n,k] (f16) = W[k,n] (f32)
// ---------------------------------------------------------------------------
__global__ void __launch_bounds__(256)
transpose_cvt_tiled_kernel(const float* __restrict__ W,
                           _Float16* __restrict__ WT, int K, int N) {
  __shared__ float tile[32][33];
  const int nt = blockIdx.x * 32;
  const int kt = blockIdx.y * 32;
  const int tx = threadIdx.x & 31;
  const int ty = threadIdx.x >> 5;     // 0..7
#pragma unroll
  for (int r = 0; r < 4; ++r)
    tile[ty + r * 8][tx] = W[(size_t)(kt + ty + r * 8) * N + nt + tx];
  __syncthreads();
#pragma unroll
  for (int r = 0; r < 4; ++r)
    WT[(size_t)(nt + ty + r * 8) * K + kt + tx] = (_Float16)tile[tx][ty + r * 8];
}

// Naive fallback for the tiny gating head (N=4 not a multiple of 32)
__global__ void transpose_cvt_kernel(const float* __restrict__ W,
                                     _Float16* __restrict__ WT, int K, int N) {
  int idx = blockIdx.x * blockDim.x + threadIdx.x;
  if (idx >= K * N) return;
  int n = idx / K;
  int k = idx - n * K;
  WT[idx] = (_Float16)W[(size_t)k * N + n];
}

// ---------------------------------------------------------------------------
// Normalize + convert: xn[b,s] = (x[b,s]-mean[s])/std[s] as f16. S=512.
// ---------------------------------------------------------------------------
__global__ void normalize_cvt_kernel(const float* __restrict__ x,
                                     const float* __restrict__ mean,
                                     const float* __restrict__ stdv,
                                     _Float16* __restrict__ xn) {
  int i = blockIdx.x * blockDim.x + threadIdx.x;   // quad index
  int s = (i * 4) & (S_SZ - 1);
  v4f xv = *(const v4f*)(x + (size_t)i * 4);
  v4f mv = *(const v4f*)(mean + s);
  v4f sv = *(const v4f*)(stdv + s);
  v4h r;
#pragma unroll
  for (int t = 0; t < 4; ++t) r[t] = (_Float16)((xv[t] - mv[t]) / sv[t]);
  *(v4h*)(xn + (size_t)i * 4) = r;
}

// ---------------------------------------------------------------------------
// GEMM + bias + tanh:  Y[B,N] = tanh(X[B,K] @ WT[N,K]^T + bias)
// Block = 256 threads = 8 waves. WG tile 128(M) x 128(N), K-step 32.
// A/B K-slabs double-buffered in LDS via global_load_async_to_lds_b128.
// LDS rows padded to 40 halves (80B) -> conflict-free 16-lane fragment reads.
// ---------------------------------------------------------------------------
#define ROW_H   40                       // halves per LDS tile row (32 + 8 pad)
#define ROW_B   (ROW_H * 2)              // 80 bytes
#define TILE_B  (128 * ROW_B)            // 10240 bytes per buffer

__global__ void __launch_bounds__(256)
gemm_tanh_kernel(const _Float16* __restrict__ X,
                 const _Float16* __restrict__ WT,
                 const float* __restrict__ bias,
                 _Float16* __restrict__ Y, int K, int N) {
  __shared__ __align__(32) _Float16 As[2][128 * ROW_H];
  __shared__ __align__(32) _Float16 Bs[2][128 * ROW_H];

  const int t     = threadIdx.x;
  const int lane  = t & 31;
  const int wid   = t >> 5;
  const int wm    = wid & 3;            // wave M index (0..3)
  const int wn    = wid >> 2;           // wave N index (0..1)
  const int mBase = blockIdx.x * 128;
  const int nBase = blockIdx.y * 128;
  const int lm    = lane & 15;
  const int hi    = lane >> 4;

  // Staging: 512 chunks of 16B per 128x32 tile; this thread owns c and c+256.
  const int c1 = t, c2 = t + 256;
  const int r1 = c1 >> 2, o1 = (c1 & 3) << 3;   // row, half-offset within slab
  const int r2 = c2 >> 2, o2 = (c2 & 3) << 3;
  const _Float16* gA1 = X  + (size_t)(mBase + r1) * K + o1;
  const _Float16* gA2 = X  + (size_t)(mBase + r2) * K + o2;
  const _Float16* gB1 = WT + (size_t)(nBase + r1) * K + o1;
  const _Float16* gB2 = WT + (size_t)(nBase + r2) * K + o2;
  const uint32_t asB = (uint32_t)(uintptr_t)&As[0][0];
  const uint32_t bsB = (uint32_t)(uintptr_t)&Bs[0][0];
  const uint32_t lA1 = asB + (uint32_t)(r1 * ROW_B + (c1 & 3) * 16);
  const uint32_t lA2 = asB + (uint32_t)(r2 * ROW_B + (c2 & 3) * 16);
  const uint32_t lB1 = bsB + (uint32_t)(r1 * ROW_B + (c1 & 3) * 16);
  const uint32_t lB2 = bsB + (uint32_t)(r2 * ROW_B + (c2 & 3) * 16);

  const int Ksteps = K >> 5;

  // Preload slab 0 into buffer 0.
  async_copy_b128(lA1, gA1);
  async_copy_b128(lA2, gA2);
  async_copy_b128(lB1, gB1);
  async_copy_b128(lB2, gB2);
  wait_async0();
  __syncthreads();

  v8f acc[2][4] = {};
  for (int ks = 0; ks < Ksteps; ++ks) {
    const int cur = ks & 1;
    if (ks + 1 < Ksteps) {               // issue async loads for next slab
      const uint32_t bo = (uint32_t)(((ks + 1) & 1) * TILE_B);
      const size_t adv = (size_t)(ks + 1) * 32;
      async_copy_b128(lA1 + bo, gA1 + adv);
      async_copy_b128(lA2 + bo, gA2 + adv);
      async_copy_b128(lB1 + bo, gB1 + adv);
      async_copy_b128(lB2 + bo, gB2 + adv);
    }

    const _Float16* ap = &As[cur][(wm * 32 + lm) * ROW_H + hi * 16];
    const _Float16* bp = &Bs[cur][(wn * 64 + lm) * ROW_H + hi * 16];
    v16h a[2], b[4];
#pragma unroll
    for (int i = 0; i < 2; ++i)
      a[i] = frag16_cont(ap + i * 16 * ROW_H);
#pragma unroll
    for (int j = 0; j < 4; ++j)
      b[j] = frag16_cont(bp + j * 16 * ROW_H);
#pragma unroll
    for (int i = 0; i < 2; ++i)
#pragma unroll
      for (int j = 0; j < 4; ++j)
        acc[i][j] = wmma_f16(a[i], b[j], acc[i][j]);

    if (ks + 1 < Ksteps) {
      wait_async0();
      __syncthreads();
    }
  }

  // Epilogue: C/D layout -> element (M = r + 8*(lane>=16), N = lane&15)
  const int m0 = mBase + wm * 32;
  const int n0 = nBase + wn * 64;
#pragma unroll
  for (int j = 0; j < 4; ++j) {
    const int n  = n0 + j * 16 + lm;
    const float bn = bias[n];
#pragma unroll
    for (int i = 0; i < 2; ++i) {
#pragma unroll
      for (int r = 0; r < 8; ++r) {
        int m = m0 + i * 16 + (hi << 3) + r;
        Y[(size_t)m * N + n] = (_Float16)fast_tanh(acc[i][j][r] + bn);
      }
    }
  }
}

// ---------------------------------------------------------------------------
// Gating head + softmax over E=4. One thread per batch row.
// ---------------------------------------------------------------------------
__global__ void __launch_bounds__(256)
gating_softmax_kernel(const _Float16* __restrict__ h,     // [B, H1G]
                      const _Float16* __restrict__ WaT,   // [E, H1G]
                      const float* __restrict__ gba,
                      float* __restrict__ wgate) {        // [B, E]
  int b = blockIdx.x * blockDim.x + threadIdx.x;
  const _Float16* hp = h + (size_t)b * H1G_SZ;
  float logit[E_SZ] = {gba[0], gba[1], gba[2], gba[3]};
  for (int k = 0; k < H1G_SZ; k += 8) {
    v8h hv = *(const v8h*)(hp + k);
#pragma unroll
    for (int e = 0; e < E_SZ; ++e) {
      v8h wv = *(const v8h*)(WaT + (size_t)e * H1G_SZ + k);
#pragma unroll
      for (int t = 0; t < 8; ++t)
        logit[e] += (float)hv[t] * (float)wv[t];
    }
  }
  float mx = fmaxf(fmaxf(logit[0], logit[1]), fmaxf(logit[2], logit[3]));
  float ex[E_SZ], s = 0.f;
#pragma unroll
  for (int e = 0; e < E_SZ; ++e) {
    ex[e] = __builtin_amdgcn_exp2f((logit[e] - mx) * 1.4426950408889634f);
    s += ex[e];
  }
  float inv = __builtin_amdgcn_rcpf(s);
#pragma unroll
  for (int e = 0; e < E_SZ; ++e) wgate[(size_t)b * E_SZ + e] = ex[e] * inv;
}

// ---------------------------------------------------------------------------
// Expert head + weighted combine:
//   out[b,:] (+)= w[b,e] * ((h1[b,:] @ eWa + eba) * scale)
// Block = 256 threads = 8 waves; each wave does 32(M) x 64(N=A).
// ---------------------------------------------------------------------------
__global__ void __launch_bounds__(256)
combine_kernel(const _Float16* __restrict__ X,      // [B, H1E]
               const _Float16* __restrict__ WT,     // [A, H1E]
               const float* __restrict__ eba_e,     // [A]
               const float* __restrict__ scale,     // [A]
               const float* __restrict__ wgate,     // [B, E]
               int e, int accum,
               float* __restrict__ out) {           // [B, A]
  const int lane = threadIdx.x & 31;
  const int wid  = threadIdx.x >> 5;
  const int m0   = blockIdx.x * 256 + wid * 32;
  const int lm   = lane & 15;
  const int K    = H1E_SZ;

  v8f acc[2][4] = {};
  for (int k0 = 0; k0 < K; k0 += 32) {
    v16h a[2], b[4];
#pragma unroll
    for (int i = 0; i < 2; ++i)
      a[i] = load_a_frag(X, K, m0 + i * 16 + lm, k0, lane);
#pragma unroll
    for (int j = 0; j < 4; ++j)
      b[j] = load_b_frag(WT, K, j * 16 + lm, k0, lane);
#pragma unroll
    for (int i = 0; i < 2; ++i)
#pragma unroll
      for (int j = 0; j < 4; ++j)
        acc[i][j] = wmma_f16(a[i], b[j], acc[i][j]);
  }

#pragma unroll
  for (int i = 0; i < 2; ++i) {
#pragma unroll
    for (int r = 0; r < 8; ++r) {
      int m = m0 + i * 16 + ((lane >> 4) << 3) + r;
      float wv = wgate[(size_t)m * E_SZ + e];
#pragma unroll
      for (int j = 0; j < 4; ++j) {
        int n = j * 16 + lm;
        float v = (acc[i][j][r] + eba_e[n]) * scale[n] * wv;
        float* op = out + (size_t)m * A_SZ + n;
        if (accum) *op += v; else *op = v;
      }
    }
  }
}

// ---------------------------------------------------------------------------
// Host orchestration
// ---------------------------------------------------------------------------
extern "C" void kernel_launch(void* const* d_in, const int* in_sizes, int n_in,
                              void* d_out, int out_size, void* d_ws, size_t ws_size,
                              hipStream_t stream) {
  (void)in_sizes; (void)n_in; (void)out_size; (void)ws_size;
  const float* x      = (const float*)d_in[0];
  const float* g_mean = (const float*)d_in[1];
  const float* g_std  = (const float*)d_in[2];
  const float* gW0    = (const float*)d_in[3];
  const float* gb0    = (const float*)d_in[4];
  const float* gW1    = (const float*)d_in[5];
  const float* gb1    = (const float*)d_in[6];
  const float* gWa    = (const float*)d_in[7];
  const float* gba    = (const float*)d_in[8];
  const float* e_mean = (const float*)d_in[9];
  const float* e_std  = (const float*)d_in[10];
  const float* eW0    = (const float*)d_in[11];
  const float* eb0    = (const float*)d_in[12];
  const float* eW1    = (const float*)d_in[13];
  const float* eb1    = (const float*)d_in[14];
  const float* eWa    = (const float*)d_in[15];
  const float* eba    = (const float*)d_in[16];
  const float* scale  = (const float*)d_in[17];
  float* out = (float*)d_out;

  // Workspace carve-up (~73 MB total)
  char* p = (char*)d_ws;
  auto carve = [&](size_t bytes) -> char* {
    char* r = p; p += (bytes + 255) & ~(size_t)255; return r;
  };
  _Float16* wtG0 = (_Float16*)carve((size_t)H0G_SZ * S_SZ   * 2);
  _Float16* wtG1 = (_Float16*)carve((size_t)H1G_SZ * H0G_SZ * 2);
  _Float16* wtGa = (_Float16*)carve((size_t)E_SZ   * H1G_SZ * 2);
  _Float16* wtE0 = (_Float16*)carve((size_t)E_SZ * H0E_SZ * S_SZ   * 2);
  _Float16* wtE1 = (_Float16*)carve((size_t)E_SZ * H1E_SZ * H0E_SZ * 2);
  _Float16* wtEa = (_Float16*)carve((size_t)E_SZ * A_SZ   * H1E_SZ * 2);
  _Float16* xn   = (_Float16*)carve((size_t)B_SZ * S_SZ   * 2);
  _Float16* bufA = (_Float16*)carve((size_t)B_SZ * H0E_SZ * 2);
  _Float16* bufB = (_Float16*)carve((size_t)B_SZ * H1E_SZ * 2);
  float*    wgt  = (float*)   carve((size_t)B_SZ * E_SZ   * 4);

  dim3 blk(256);

  // --- weight transpose + f32->f16 conversion ---
  transpose_cvt_tiled_kernel<<<dim3(H0G_SZ / 32, S_SZ / 32), blk, 0, stream>>>(
      gW0, wtG0, S_SZ, H0G_SZ);
  transpose_cvt_tiled_kernel<<<dim3(H1G_SZ / 32, H0G_SZ / 32), blk, 0, stream>>>(
      gW1, wtG1, H0G_SZ, H1G_SZ);
  transpose_cvt_kernel<<<(H1G_SZ * E_SZ + 255) / 256, blk, 0, stream>>>(
      gWa, wtGa, H1G_SZ, E_SZ);
  for (int e = 0; e < E_SZ; ++e) {
    transpose_cvt_tiled_kernel<<<dim3(H0E_SZ / 32, S_SZ / 32), blk, 0, stream>>>(
        eW0 + (size_t)e * S_SZ * H0E_SZ, wtE0 + (size_t)e * H0E_SZ * S_SZ, S_SZ, H0E_SZ);
    transpose_cvt_tiled_kernel<<<dim3(H1E_SZ / 32, H0E_SZ / 32), blk, 0, stream>>>(
        eW1 + (size_t)e * H0E_SZ * H1E_SZ, wtE1 + (size_t)e * H1E_SZ * H0E_SZ, H0E_SZ, H1E_SZ);
    transpose_cvt_tiled_kernel<<<dim3(A_SZ / 32, H1E_SZ / 32), blk, 0, stream>>>(
        eWa + (size_t)e * H1E_SZ * A_SZ, wtEa + (size_t)e * A_SZ * H1E_SZ, H1E_SZ, A_SZ);
  }

  // --- gating network ---
  normalize_cvt_kernel<<<(B_SZ * S_SZ / 4) / 256, blk, 0, stream>>>(x, g_mean, g_std, xn);
  gemm_tanh_kernel<<<dim3(B_SZ / 128, H0G_SZ / 128), blk, 0, stream>>>(
      xn, wtG0, gb0, bufA, S_SZ, H0G_SZ);
  gemm_tanh_kernel<<<dim3(B_SZ / 128, H1G_SZ / 128), blk, 0, stream>>>(
      bufA, wtG1, gb1, bufB, H0G_SZ, H1G_SZ);
  gating_softmax_kernel<<<B_SZ / 256, blk, 0, stream>>>(bufB, wtGa, gba, wgt);

  // --- experts, sequential, reusing bufA/bufB; combine accumulates into out ---
  for (int e = 0; e < E_SZ; ++e) {
    normalize_cvt_kernel<<<(B_SZ * S_SZ / 4) / 256, blk, 0, stream>>>(
        x, e_mean + (size_t)e * S_SZ, e_std + (size_t)e * S_SZ, xn);
    gemm_tanh_kernel<<<dim3(B_SZ / 128, H0E_SZ / 128), blk, 0, stream>>>(
        xn, wtE0 + (size_t)e * H0E_SZ * S_SZ, eb0 + (size_t)e * H0E_SZ,
        bufA, S_SZ, H0E_SZ);
    gemm_tanh_kernel<<<dim3(B_SZ / 128, H1E_SZ / 128), blk, 0, stream>>>(
        bufA, wtE1 + (size_t)e * H1E_SZ * H0E_SZ, eb1 + (size_t)e * H1E_SZ,
        bufB, H0E_SZ, H1E_SZ);
    combine_kernel<<<B_SZ / 256, blk, 0, stream>>>(
        bufB, wtEa + (size_t)e * A_SZ * H1E_SZ, eba + (size_t)e * A_SZ,
        scale, wgt, e, (e > 0) ? 1 : 0, out);
  }
}
